// _LongcatFlashRouter_85787676770797
// MI455X (gfx1250) — matrix operand order repaired and checked by reading, not compiled, not run on hardware
//
#include <hip/hip_runtime.h>
#include <hip/hip_bf16.h>
#include <stdint.h>

#define NUM_TOKENS   8192
#define HIDDEN       4096
#define NUM_EXPERTS  512
#define TOPK         12
#define ROUTED_SCALE 2.5f
#define KCHUNKS      (HIDDEN / 32)   // 128
#define TILES_PER_WAVE 4

typedef __attribute__((ext_vector_type(16))) __bf16 bf16x16;
typedef __attribute__((ext_vector_type(16))) float  f32x16;
typedef __attribute__((ext_vector_type(8)))  float  f32x8;

union Frag {
  bf16x16 v;
  unsigned u[8];
  uint4    q[2];
};

// 16x fp32 -> bf16 (RNE); lowers to v_cvt_pk_bf16_f32 on gfx1250 (verified).
static __device__ __forceinline__ bf16x16 cvt16(const f32x16& f) {
  return __builtin_convertvector(f, bf16x16);
}

static __device__ __forceinline__ bf16x16 make_afrag(float4 a0, float4 a1,
                                                     float4 a2, float4 a3) {
  f32x16 ff;
  ff[0]  = a0.x; ff[1]  = a0.y; ff[2]  = a0.z; ff[3]  = a0.w;
  ff[4]  = a1.x; ff[5]  = a1.y; ff[6]  = a1.z; ff[7]  = a1.w;
  ff[8]  = a2.x; ff[9]  = a2.y; ff[10] = a2.z; ff[11] = a2.w;
  ff[12] = a3.x; ff[13] = a3.y; ff[14] = a3.z; ff[15] = a3.w;
  return cvt16(ff);
}

// Load one K-chunk: 4x float4 of A plus 4 register-staged B fragments.
#define LOAD_CHUNK(KC, A0, A1, A2, A3, BB)                                   \
  do {                                                                       \
    const float4* _ap = (const float4*)(Abase + (size_t)(KC) * 32);          \
    A0 = _ap[0]; A1 = _ap[1]; A2 = _ap[2]; A3 = _ap[3];                      \
    _Pragma("unroll")                                                        \
    for (int _j = 0; _j < TILES_PER_WAVE; ++_j) {                            \
      const uint4* _src = bwave + ((size_t)_j * KCHUNKS + (KC)) * 64;        \
      BB[_j].q[0] = _src[0];                                                 \
      BB[_j].q[1] = _src[1];                                                 \
    }                                                                        \
  } while (0)

// Convert A and run the 4-tile WMMA chain for one K-chunk.
#define COMPUTE_CHUNK(A0, A1, A2, A3, BB)                                    \
  do {                                                                       \
    Frag _A;                                                                 \
    _A.v = make_afrag(A0, A1, A2, A3);                                       \
    _Pragma("unroll")                                                        \
    for (int _j = 0; _j < TILES_PER_WAVE; ++_j)                              \
      accs[_j] = __builtin_amdgcn_wmma_f32_16x16x32_bf16(                    \
          false, _A.v, false, BB[_j].v, (short)0, accs[_j], false, false);   \
  } while (0)

// ---------------------------------------------------------------------------
// Kernel 1: convert classifier weights fp32 -> bf16, pre-swizzled into the
// WMMA B-fragment layout:  bp[etile][kchunk][lane][8 dwords]
//   lane 0-15 : expert = etile*16+lane,    K = kc*32 + 0..15  (2 per dword)
//   lane 16-31: expert = etile*16+lane-16, K = kc*32 + 16..31
// ---------------------------------------------------------------------------
__global__ __launch_bounds__(256) void prepack_weights(const float* __restrict__ w,
                                                       unsigned* __restrict__ bp) {
  const int t    = blockIdx.x * 256 + threadIdx.x;   // 32*128*32 = 131072 threads
  const int lane = t & 31;
  const int kc   = (t >> 5) & (KCHUNKS - 1);
  const int tile = t >> 12;                          // 0..31
  const int expert = tile * 16 + (lane & 15);
  const int kstart = kc * 32 + ((lane >> 4) << 4);
  const float4* src = (const float4*)(w + (size_t)expert * HIDDEN + kstart);
  float4 f0 = src[0], f1 = src[1], f2 = src[2], f3 = src[3];
  Frag o;
  o.v = make_afrag(f0, f1, f2, f3);
  uint4* dst = (uint4*)(bp + (size_t)t * 8);
  dst[0] = o.q[0]; dst[1] = o.q[1];
}

// ---------------------------------------------------------------------------
// Kernel 2: fused GEMM (WMMA bf16) + softmax + biased top-12.
// Block = 32 tokens x 512 experts. 512 threads = 16 waves.
// Wave w: token-tile mt = w&1, expert tiles (w>>1)*4 .. +3.
// Manual X/Y ping-pong double-buffering: no register rotation, loads for the
// next chunk are always in flight behind the current chunk's WMMA chain.
// ---------------------------------------------------------------------------
__global__ __launch_bounds__(512) void router_fused(const float* __restrict__ hs,
                                                    const unsigned* __restrict__ bp,
                                                    const float* __restrict__ bias,
                                                    float* __restrict__ w_out,
                                                    int* __restrict__ e_out) {
  __shared__ float sm[32 * NUM_EXPERTS];   // 64 KB logits

  const int tid  = threadIdx.x;
  const int lane = tid & 31;
  const int wv   = tid >> 5;      // 0..15
  const int mt   = wv & 1;
  const int eg   = wv >> 1;       // 0..7
  const int blk  = blockIdx.x;
  const int l16  = lane & 15;
  const int half = lane >> 4;

  // A-fragment source: contiguous k order (k = 16*half + s), same as packed B.
  //   lane 0-15 : M=lane,    K = kc*32 + 0..15
  //   lane 16-31: M=lane-16, K = kc*32 + 16..31
  const float* Abase = hs + (size_t)(blk * 32 + mt * 16 + l16) * HIDDEN + half * 16;
  // Per-wave B base: tiles eg*4..eg*4+3 live at consecutive 128KB strides.
  const uint4* bwave = (const uint4*)bp +
                       ((size_t)(eg * TILES_PER_WAVE) * KCHUNKS * 32 + lane) * 2;

  const f32x8 zero = {0.f, 0.f, 0.f, 0.f, 0.f, 0.f, 0.f, 0.f};
  f32x8 accs[TILES_PER_WAVE];
#pragma unroll
  for (int j = 0; j < TILES_PER_WAVE; ++j) accs[j] = zero;

  // ---- prologue: chunk 0 -> X ----
  float4 xa0, xa1, xa2, xa3;
  Frag XB[TILES_PER_WAVE];
  LOAD_CHUNK(0, xa0, xa1, xa2, xa3, XB);

  // ---- main loop: 63 iterations, two chunks each (chunks 0..125),
  //      leaves X holding chunk 126 ----
  for (int i = 0; i < (KCHUNKS - 2) / 2; ++i) {
    const int k0 = 2 * i;
    float4 ya0, ya1, ya2, ya3;
    Frag YB[TILES_PER_WAVE];
    LOAD_CHUNK(k0 + 1, ya0, ya1, ya2, ya3, YB);      // prefetch odd chunk
    COMPUTE_CHUNK(xa0, xa1, xa2, xa3, XB);           // compute even chunk
    LOAD_CHUNK(k0 + 2, xa0, xa1, xa2, xa3, XB);      // prefetch next even
    COMPUTE_CHUNK(ya0, ya1, ya2, ya3, YB);           // compute odd chunk
  }

  // ---- epilogue: chunks 126 (in X) and 127 ----
  {
    float4 ya0, ya1, ya2, ya3;
    Frag YB[TILES_PER_WAVE];
    LOAD_CHUNK(KCHUNKS - 1, ya0, ya1, ya2, ya3, YB);
    COMPUTE_CHUNK(xa0, xa1, xa2, xa3, XB);
    COMPUTE_CHUNK(ya0, ya1, ya2, ya3, YB);
  }

  // C layout: VGPR r, lane<16 -> M=r, lane>=16 -> M=r+8; N = lane&15.
#pragma unroll
  for (int j = 0; j < TILES_PER_WAVE; ++j) {
    const int ecol = (eg * TILES_PER_WAVE + j) * 16 + l16;
    const int trow = mt * 16 + half * 8;
#pragma unroll
    for (int r = 0; r < 8; ++r)
      sm[(trow + r) * NUM_EXPERTS + ecol] = accs[j][r];
  }
  __syncthreads();

  // ---- Phase 2: per-token softmax + biased top-12 (wave per token) ----
  for (int tk = 0; tk < 2; ++tk) {
    const int t = wv * 2 + tk;            // local token 0..31
    float lg[16], bv[16], sc[16];
#pragma unroll
    for (int i = 0; i < 16; ++i) lg[i] = sm[t * NUM_EXPERTS + lane + 32 * i];
#pragma unroll
    for (int i = 0; i < 16; ++i) bv[i] = bias[lane + 32 * i];

    float mx = lg[0];
#pragma unroll
    for (int i = 1; i < 16; ++i) mx = fmaxf(mx, lg[i]);
#pragma unroll
    for (int off = 16; off >= 1; off >>= 1) mx = fmaxf(mx, __shfl_xor(mx, off, 32));

    float sum = 0.f;
#pragma unroll
    for (int i = 0; i < 16; ++i) { lg[i] = __expf(lg[i] - mx); sum += lg[i]; }
#pragma unroll
    for (int off = 16; off >= 1; off >>= 1) sum += __shfl_xor(sum, off, 32);
    const float inv = 1.0f / sum;

    // biased selection score; unbiased prob recovered later as (score - bias)
#pragma unroll
    for (int i = 0; i < 16; ++i) sc[i] = lg[i] * inv + bv[i];

    const size_t tok = (size_t)blk * 32 + t;
    for (int k = 0; k < TOPK; ++k) {
      // lane-local argmax (ascending i => keep-first tie-break = lower index)
      float bs = sc[0]; int bi = lane;
#pragma unroll
      for (int i = 1; i < 16; ++i) {
        float v = sc[i]; int vi = lane + 32 * i;
        if (v > bs) { bs = v; bi = vi; }
      }
      // wave argmax with lower-index tie-break
#pragma unroll
      for (int off = 16; off >= 1; off >>= 1) {
        float os = __shfl_xor(bs, off, 32);
        int   oi = __shfl_xor(bi, off, 32);
        if (os > bs || (os == bs && oi < bi)) { bs = os; bi = oi; }
      }
      const int slot  = bi >> 5;   // uniform across wave
      const int owner = bi & 31;   // uniform across wave
      float bsel = 0.f;
#pragma unroll
      for (int i = 0; i < 16; ++i) bsel = (i == slot) ? bv[i] : bsel;
      const float bbest = __shfl(bsel, owner, 32);
      if (lane == 0) {
        w_out[tok * TOPK + k] = (bs - bbest) * ROUTED_SCALE;
        e_out[tok * TOPK + k] = bi;
      }
#pragma unroll
      for (int i = 0; i < 16; ++i)
        if (i == slot && lane == owner) sc[i] = -__builtin_inff();
    }
  }
}

// ---------------------------------------------------------------------------
extern "C" void kernel_launch(void* const* d_in, const int* in_sizes, int n_in,
                              void* d_out, int out_size, void* d_ws, size_t ws_size,
                              hipStream_t stream) {
  (void)in_sizes; (void)n_in; (void)out_size; (void)ws_size;
  const float* hs   = (const float*)d_in[0];
  const float* cw   = (const float*)d_in[1];
  const float* bias = (const float*)d_in[2];
  float* w_out = (float*)d_out;
  int*   e_out = (int*)((float*)d_out + (size_t)NUM_TOKENS * TOPK);
  unsigned* bp = (unsigned*)d_ws;   // needs 4 MiB scratch for packed bf16 weights

  // 32 etiles * 128 kchunks * 32 lanes = 131072 threads
  prepack_weights<<<512, 256, 0, stream>>>(cw, bp);
  router_fused<<<NUM_TOKENS / 32, 512, 0, stream>>>(hs, bp, bias, w_out, e_out);
}
